// GraphValueModel_74775380624043
// MI455X (gfx1250) — matrix operand order repaired and checked by reading, not compiled, not run on hardware
//
#include <hip/hip_runtime.h>
#include <hip/hip_bf16.h>

// GraphValueModel: B=32, N=192, MID=64 interaction network.
// Messenger MLP over all pairs dominates (~20 GFLOP) -> WMMA f16/f32.
// GEMMs are computed transposed (weights = A-matrix, activations = B-matrix)
// so D tiles store back to LDS as a single packed ds_store_b128 per tile.

#define BB   32
#define NN   192
#define MIDF 64
#define HSTR 72   // LDS row stride in halves (144 B: 16B-aligned, depadded banks)

typedef __attribute__((ext_vector_type(16))) _Float16 v16h;
typedef __attribute__((ext_vector_type(8)))  _Float16 v8h;
typedef __attribute__((ext_vector_type(8)))  float    v8f;

// ---------------------------------------------------------------------------
// Fragment helpers (wave32), contiguous k order k = (lane>>4)*16 + e in A and B.
// ---------------------------------------------------------------------------

// A-matrix (16x32 f16, M x K) = transposed weight tile W^T[m=out feat][k=in].
// Source W is row-major [64 in][64 out]: A[m][k] = W[k][m].
// lane&15 = row m. element e <-> K = (lane>>4)*16 + e (matches B fragment).
__device__ inline v16h load_wT_a_frag(const float* __restrict__ W,
                                      int mTile, int kTile, int lane) {
    const int m  = mTile * 16 + (lane & 15);
    const int hi = lane >> 4;
    const int k0 = kTile * 32 + hi * 16;       // elems 0-7
    const int k1 = k0 + 8;                     // elems 8-15
    v16h a;
#pragma unroll
    for (int e = 0; e < 8; ++e) {
        a[e]     = (_Float16)W[(k0 + e) * MIDF + m];
        a[8 + e] = (_Float16)W[(k1 + e) * MIDF + m];
    }
    return a;
}

// B-matrix (32x16 f16, K x N) = activation tile h^T: column n = node j.
// lane&15 = column (node j); element e <-> K = (lane>>4)*16 + e.
// Row-major h[j][feat] makes the 16 halves contiguous -> 2x ds_load_b128.
__device__ inline v16h load_h_b_frag(const _Float16* __restrict__ h,
                                     int jTile, int kTile, int lane) {
    const int j = jTile * 16 + (lane & 15);
    const _Float16* p = h + j * HSTR + kTile * 32 + (lane >> 4) * 16;
    v8h lo = *(const v8h*)(p);        // 16B aligned
    v8h hv = *(const v8h*)(p + 8);
    v16h b;
#pragma unroll
    for (int e = 0; e < 8; ++e) { b[e] = lo[e]; b[8 + e] = hv[e]; }
    return b;
}

// D tile (16x16 f32): lane&15 = column (node j); element v <-> output
// feature mTile*16 + (lane>>4)*8 + v. Bias+ReLU fused, packed single store.
__device__ inline void store_d_relu_lds(v8f c, _Float16* __restrict__ h_out,
                                        int mTile, int jTile, v8f biasv,
                                        int lane) {
    const int j        = jTile * 16 + (lane & 15);
    const int featbase = mTile * 16 + (lane >> 4) * 8;
    v8h p;
#pragma unroll
    for (int v = 0; v < 8; ++v) {
        float r = c[v] + biasv[v];
        p[v] = (_Float16)(r > 0.f ? r : 0.f);
    }
    *(v8h*)(h_out + j * HSTR + featbase) = p;   // ds_store_b128
}

// One transposed 64x64 layer over 192 nodes: h_out = relu(h_in @ W + b).
// Wave owns output-feature tile mTile = wave&3 and 6 node tiles.
__device__ inline void gemm_layer(const _Float16* __restrict__ h_in,
                                  _Float16* __restrict__ h_out,
                                  v16h wk0, v16h wk1, v8f biasv,
                                  int wave, int lane) {
    const int mTile  = wave & 3;
    const int jStart = (wave >> 2) * 6;
#pragma unroll
    for (int jt = jStart; jt < jStart + 6; ++jt) {
        v8f c = {};
        v16h b0 = load_h_b_frag(h_in, jt, 0, lane);
        c = __builtin_amdgcn_wmma_f32_16x16x32_f16(false, wk0, false, b0,
                                                   (short)0, c, false, false);
        v16h b1 = load_h_b_frag(h_in, jt, 1, lane);
        c = __builtin_amdgcn_wmma_f32_16x16x32_f16(false, wk1, false, b1,
                                                   (short)0, c, false, false);
        store_d_relu_lds(c, h_out, mTile, jt, biasv, lane);
    }
}

__device__ inline v8f load_bias_vec(const float* __restrict__ bias, int mTile,
                                    int lane) {
    const int base = mTile * 16 + (lane >> 4) * 8;
    v8f bv;
#pragma unroll
    for (int v = 0; v < 8; ++v) bv[v] = bias[base + v];
    return bv;
}

// ---------------------------------------------------------------------------
// Kernel 1: per-node first-layer factorization.
// A[b,i,k] = x0*mW0[0,k] + x1*mW0[1,k]          (receiver half)
// C[b,j,k] = x0*mW0[2,k] + x1*mW0[3,k] + mb0[k] (sender half + bias)
// ---------------------------------------------------------------------------
__global__ __launch_bounds__(256) void pre_kernel(
    const float* __restrict__ obs, const float* __restrict__ mW0,
    const float* __restrict__ mb0, float* __restrict__ Af,
    float* __restrict__ Cf) {
    int idx = blockIdx.x * 256 + threadIdx.x;      // (b*192+n)*64 + k
    if (idx >= BB * NN * MIDF) return;
    int k  = idx & (MIDF - 1);
    int bn = idx >> 6;
    int n  = bn % NN;
    float x0 = obs[bn];
    float x1 = (float)n / (float)NN;
    Af[idx] = x0 * mW0[k]            + x1 * mW0[MIDF + k];
    Cf[idx] = x0 * mW0[2 * MIDF + k] + x1 * mW0[3 * MIDF + k] + mb0[k];
}

// ---------------------------------------------------------------------------
// Kernel 2: messenger. One workgroup per (b,i). h1 = relu(A_i + C_j) in LDS,
// two WMMA layers, then msg[b,i] = (sum_j h3)·W3 + N*b3.
// ---------------------------------------------------------------------------
__global__ __launch_bounds__(256) void msg_kernel(
    const float* __restrict__ Af, const float* __restrict__ Cf,
    const float* __restrict__ mW1, const float* __restrict__ mb1,
    const float* __restrict__ mW2, const float* __restrict__ mb2,
    const float* __restrict__ mW3, const float* __restrict__ mb3,
    float* __restrict__ msg_out) {
    __shared__ __attribute__((aligned(16))) _Float16 hA[NN * HSTR];
    __shared__ __attribute__((aligned(16))) _Float16 hB[NN * HSTR];
    __shared__ float red[4 * MIDF];

    const int bi   = blockIdx.x;           // b*192 + i
    const int b    = bi / NN;
    const int tid  = threadIdx.x;
    const int lane = tid & 31;
    const int wave = tid >> 5;
    const int mTile = wave & 3;

    // Preload this wave's transposed-weight A fragments + bias vectors.
    v16h w1k0 = load_wT_a_frag(mW1, mTile, 0, lane);
    v16h w1k1 = load_wT_a_frag(mW1, mTile, 1, lane);
    v16h w2k0 = load_wT_a_frag(mW2, mTile, 0, lane);
    v16h w2k1 = load_wT_a_frag(mW2, mTile, 1, lane);
    v8f bias1 = load_bias_vec(mb1, mTile, lane);
    v8f bias2 = load_bias_vec(mb2, mTile, lane);

    // h1[j,k] = relu(A[b,i,k] + C[b,j,k])
    const float* Ai = Af + (size_t)bi * MIDF;
    const float* Cb = Cf + (size_t)b * NN * MIDF;
    for (int e = tid; e < NN * MIDF; e += 256) {
        int j = e >> 6, k = e & (MIDF - 1);
        float v = Ai[k] + Cb[e];
        hA[j * HSTR + k] = (_Float16)(v > 0.f ? v : 0.f);
    }
    __syncthreads();

    gemm_layer(hA, hB, w1k0, w1k1, bias1, wave, lane);   // layer 1
    __syncthreads();
    gemm_layer(hB, hA, w2k0, w2k1, bias2, wave, lane);   // layer 2
    __syncthreads();

    // msg = (sum_j h3[j,:]) · W3 + N*b3   (final linear layer folded in)
    const int k = tid & (MIDF - 1), part = tid >> 6;
    float s = 0.f;
    for (int j = part * 48; j < part * 48 + 48; ++j)
        s += (float)hA[j * HSTR + k];
    red[part * MIDF + k] = s;
    __syncthreads();
    if (tid < MIDF) {
        float t = red[tid] + red[MIDF + tid] + red[2 * MIDF + tid] +
                  red[3 * MIDF + tid];
        red[tid] = t * mW3[tid];
    }
    __syncthreads();
    if (tid == 0) {
        float m = 0.f;
        for (int kk = 0; kk < MIDF; ++kk) m += red[kk];
        msg_out[bi] = m + (float)NN * mb3[0];
    }
}

// ---------------------------------------------------------------------------
// Kernel 3: updater. One workgroup per batch b. h = [x0,x1,msg] -> 3->64 in
// VALU, two WMMA layers, final layer folded into the sum over nodes.
// ---------------------------------------------------------------------------
__global__ __launch_bounds__(256) void upd_kernel(
    const float* __restrict__ obs, const float* __restrict__ msg_in,
    const float* __restrict__ uW0, const float* __restrict__ ub0,
    const float* __restrict__ uW1, const float* __restrict__ ub1,
    const float* __restrict__ uW2, const float* __restrict__ ub2,
    const float* __restrict__ uW3, const float* __restrict__ ub3,
    float* __restrict__ out) {
    __shared__ __attribute__((aligned(16))) _Float16 hA[NN * HSTR];
    __shared__ __attribute__((aligned(16))) _Float16 hB[NN * HSTR];
    __shared__ float red[4 * MIDF];

    const int b    = blockIdx.x;
    const int tid  = threadIdx.x;
    const int lane = tid & 31;
    const int wave = tid >> 5;
    const int mTile = wave & 3;

    v16h w1k0 = load_wT_a_frag(uW1, mTile, 0, lane);
    v16h w1k1 = load_wT_a_frag(uW1, mTile, 1, lane);
    v16h w2k0 = load_wT_a_frag(uW2, mTile, 0, lane);
    v16h w2k1 = load_wT_a_frag(uW2, mTile, 1, lane);
    v8f bias1 = load_bias_vec(ub1, mTile, lane);
    v8f bias2 = load_bias_vec(ub2, mTile, lane);

    // h1[i,k] = relu(x0*uW0[0,k] + x1*uW0[1,k] + msg*uW0[2,k] + ub0[k])
    for (int e = tid; e < NN * MIDF; e += 256) {
        int i = e >> 6, k = e & (MIDF - 1);
        float x0 = obs[b * NN + i];
        float x1 = (float)i / (float)NN;
        float m  = msg_in[b * NN + i];
        float v = x0 * uW0[k] + x1 * uW0[MIDF + k] + m * uW0[2 * MIDF + k] +
                  ub0[k];
        hA[i * HSTR + k] = (_Float16)(v > 0.f ? v : 0.f);
    }
    __syncthreads();

    gemm_layer(hA, hB, w1k0, w1k1, bias1, wave, lane);
    __syncthreads();
    gemm_layer(hB, hA, w2k0, w2k1, bias2, wave, lane);
    __syncthreads();

    // out[b] = (sum_i h3[i,:]) · uW3 + N*ub3
    const int k = tid & (MIDF - 1), part = tid >> 6;
    float s = 0.f;
    for (int i = part * 48; i < part * 48 + 48; ++i)
        s += (float)hA[i * HSTR + k];
    red[part * MIDF + k] = s;
    __syncthreads();
    if (tid < MIDF) {
        float t = red[tid] + red[MIDF + tid] + red[2 * MIDF + tid] +
                  red[3 * MIDF + tid];
        red[tid] = t * uW3[tid];
    }
    __syncthreads();
    if (tid == 0) {
        float m = 0.f;
        for (int kk = 0; kk < MIDF; ++kk) m += red[kk];
        out[b] = m + (float)NN * ub3[0];
    }
}

// ---------------------------------------------------------------------------
extern "C" void kernel_launch(void* const* d_in, const int* in_sizes, int n_in,
                              void* d_out, int out_size, void* d_ws,
                              size_t ws_size, hipStream_t stream) {
    const float* obs = (const float*)d_in[0];
    const float* mW0 = (const float*)d_in[1];
    const float* mb0 = (const float*)d_in[2];
    const float* mW1 = (const float*)d_in[3];
    const float* mb1 = (const float*)d_in[4];
    const float* mW2 = (const float*)d_in[5];
    const float* mb2 = (const float*)d_in[6];
    const float* mW3 = (const float*)d_in[7];
    const float* mb3 = (const float*)d_in[8];
    const float* uW0 = (const float*)d_in[9];
    const float* ub0 = (const float*)d_in[10];
    const float* uW1 = (const float*)d_in[11];
    const float* ub1 = (const float*)d_in[12];
    const float* uW2 = (const float*)d_in[13];
    const float* ub2 = (const float*)d_in[14];
    const float* uW3 = (const float*)d_in[15];
    const float* ub3 = (const float*)d_in[16];

    float* Af  = (float*)d_ws;                         // [32*192*64]
    float* Cf  = Af + (size_t)BB * NN * MIDF;          // [32*192*64]
    float* msg = Cf + (size_t)BB * NN * MIDF;          // [32*192]

    const int pre_elems = BB * NN * MIDF;
    pre_kernel<<<(pre_elems + 255) / 256, 256, 0, stream>>>(obs, mW0, mb0, Af,
                                                            Cf);
    msg_kernel<<<BB * NN, 256, 0, stream>>>(Af, Cf, mW1, mb1, mW2, mb2, mW3,
                                            mb3, msg);
    upd_kernel<<<BB, 256, 0, stream>>>(obs, msg, uW0, ub0, uW1, ub1, uW2, ub2,
                                       uW3, ub3, (float*)d_out);
}
